// MultiHeadAttention_2491081031948
// MI455X (gfx1250) — matrix operand order repaired and checked by reading, not compiled, not run on hardware
//
#include <hip/hip_runtime.h>
#include <hip/hip_bf16.h>
#include <stddef.h>

// MI455X / gfx1250, wave32. Flash-attention MHA, bf16 WMMA (v_wmma_f32_16x16x32_bf16)
// + async global->LDS copies (ASYNCcnt) + global_prefetch_b8 streaming hints.
// Workspace layout (needs >= 32 MiB):
//   [ 0MiB) Qh  bf16 [B,H,L,64]
//   [ 8MiB) Kh  bf16 [B,H,L,64]
//   [16MiB) Vh  bf16 [B,H,L,64]
//   [24MiB) Ctx bf16 [B,L,D]

#define D_MODEL 1024
#define L_SEQ   2048
#define NHEADS  16
#define DHEAD   64
#define NEGBIG  (-10000000.0f)

typedef __bf16 bf16_t;
typedef __attribute__((ext_vector_type(16))) __bf16 v16bf;
typedef __attribute__((ext_vector_type(8)))  float  v8f;
typedef __attribute__((ext_vector_type(4)))  float  f32x4;

union FragU { v16bf v; bf16_t e[16]; f32x4 q[2]; };
union Pack4 { bf16_t e[4]; unsigned long long u; };
union Pack8 { f32x4 q; bf16_t e[8]; };

#if __has_builtin(__builtin_amdgcn_global_load_async_to_lds_b128) && \
    __has_builtin(__builtin_amdgcn_s_wait_asynccnt)
#define HAVE_ASYNC_LDS 1
#else
#define HAVE_ASYNC_LDS 0
#endif

#define AS1 __attribute__((address_space(1)))
#define AS3 __attribute__((address_space(3)))

// exact pointee type expected by the async-to-LDS builtins (per hipcc diagnostic)
typedef int vsi4 __attribute__((vector_size(4 * sizeof(int))));

// 16-byte async copy global -> LDS (per-lane), tracked by ASYNCcnt.
static __device__ inline void async_cp_b128(void* lds_dst, const void* gsrc) {
#if HAVE_ASYNC_LDS
  __builtin_amdgcn_global_load_async_to_lds_b128(
      (AS1 vsi4*)gsrc, (AS3 vsi4*)lds_dst, 0, 0);
#endif
}

static __device__ inline void async_wait0() {
#if HAVE_ASYNC_LDS
  __builtin_amdgcn_s_wait_asynccnt(0);
#endif
}

static __device__ inline v8f wmma_bf16(v16bf a, v16bf b, v8f c) {
  // D = A(16x32 bf16) * B(32x16 bf16) + C(16x16 f32)
  return __builtin_amdgcn_wmma_f32_16x16x32_bf16(
      /*neg_a=*/false, a, /*neg_b=*/false, b,
      /*c_mod=*/(short)0, c, /*reuse_a=*/false, /*reuse_b=*/false);
}

// A-fragment (16x32, bf16) from row-major LDS tile; per-lane:
//   m = lane&15, half = lane>>4, element j -> [m][16*(j>>3) + 8*half + (j&7)]
template<int STRIDE>
static __device__ inline v16bf load_a_frag(const bf16_t* base, int lane) {
  const int m = lane & 15, half = lane >> 4;
  const bf16_t* p = base + m * STRIDE + 8 * half;
  FragU f;
  f.q[0] = *reinterpret_cast<const f32x4*>(p);       // k = 8h .. 8h+7
  f.q[1] = *reinterpret_cast<const f32x4*>(p + 16);  // k = 16+8h .. 16+8h+7
  return f.v;
}

// B-fragment (32x16, bf16) where B[k][n] == Rows[n][k] (row-major rows, contiguous k):
//   n = lane&15, half = lane>>4, element j -> Rows[n][16*(j>>3) + 8*half + (j&7)]
template<int STRIDE>
static __device__ inline v16bf load_b_frag(const bf16_t* base, int lane) {
  const int n = lane & 15, half = lane >> 4;
  const bf16_t* p = base + n * STRIDE + 8 * half;
  FragU f;
  f.q[0] = *reinterpret_cast<const f32x4*>(p);
  f.q[1] = *reinterpret_cast<const f32x4*>(p + 16);
  return f.v;
}

// ---------------------------------------------------------------------------
// Kernel 1: Y = X(fp32) @ W(fp32)^T + bias, output bf16 remapped to [B,H,L,64]
// Block tile 128x128, K-step 32, 8 waves (4 in M x 2 in N), wave tile 32x64.
// ---------------------------------------------------------------------------
__global__ __launch_bounds__(256) void qkv_gemm_kernel(
    const float* __restrict__ X,     // [4096,1024]
    const float* __restrict__ W,     // [1024,1024]
    const float* __restrict__ bias,  // [1024]
    bf16_t* __restrict__ Out)        // [B,H,L,64] bf16
{
  constexpr int LDT = 40;  // 32 + 8 pad (80B rows: 16B aligned, conflict-free)
  __shared__ bf16_t As[128 * LDT];
  __shared__ bf16_t Bs[128 * LDT];

  const int tid  = threadIdx.x;
  const int lane = tid & 31;
  const int wave = tid >> 5;
  const int wm   = wave & 3;   // 0..3
  const int wn   = wave >> 2;  // 0..1
  const int m0   = blockIdx.x * 128;
  const int n0   = blockIdx.y * 128;

  v8f acc[2][4];
#pragma unroll
  for (int i = 0; i < 2; ++i)
#pragma unroll
    for (int j = 0; j < 4; ++j) acc[i][j] = (v8f){};

  for (int k0 = 0; k0 < D_MODEL; k0 += 32) {
    // cooperative load + fp32->bf16 convert of A(128x32) and B(=W rows, 128x32)
#pragma unroll
    for (int i = 0; i < 4; ++i) {
      const int idx = tid + i * 256;      // 0..1023
      const int r   = idx >> 3;           // 0..127
      const int c   = (idx & 7) * 4;      // 0,4,..,28
      f32x4 a4 = *reinterpret_cast<const f32x4*>(X + (size_t)(m0 + r) * D_MODEL + k0 + c);
      f32x4 b4 = *reinterpret_cast<const f32x4*>(W + (size_t)(n0 + r) * D_MODEL + k0 + c);
      Pack4 pa, pb;
#pragma unroll
      for (int e = 0; e < 4; ++e) { pa.e[e] = (bf16_t)a4[e]; pb.e[e] = (bf16_t)b4[e]; }
      *reinterpret_cast<unsigned long long*>(&As[r * LDT + c]) = pa.u;
      *reinterpret_cast<unsigned long long*>(&Bs[r * LDT + c]) = pb.u;
    }
    // prefetch next K-tile (one 128B cacheline per row) -> global_prefetch_b8
    const int pk = k0 + 32;
    if (pk < D_MODEL) {
      if (tid < 128) __builtin_prefetch(X + (size_t)(m0 + tid) * D_MODEL + pk, 0, 0);
      else           __builtin_prefetch(W + (size_t)(n0 + tid - 128) * D_MODEL + pk, 0, 0);
    }
    __syncthreads();

    v16bf af[2], bfv[4];
#pragma unroll
    for (int mt = 0; mt < 2; ++mt)
      af[mt] = load_a_frag<LDT>(&As[(wm * 32 + mt * 16) * LDT], lane);
#pragma unroll
    for (int nt = 0; nt < 4; ++nt)
      bfv[nt] = load_b_frag<LDT>(&Bs[(wn * 64 + nt * 16) * LDT], lane);
#pragma unroll
    for (int mt = 0; mt < 2; ++mt)
#pragma unroll
      for (int nt = 0; nt < 4; ++nt)
        acc[mt][nt] = wmma_bf16(af[mt], bfv[nt], acc[mt][nt]);
    __syncthreads();
  }

  // epilogue: + bias, write bf16 to head-major [B,H,L,64]
  const int nl = lane & 15, half = lane >> 4;
#pragma unroll
  for (int mt = 0; mt < 2; ++mt) {
#pragma unroll
    for (int nt = 0; nt < 4; ++nt) {
      const int o  = n0 + wn * 64 + nt * 16 + nl;  // output feature
      const int h  = o >> 6, dh = o & 63;
      const float bv = bias[o];
#pragma unroll
      for (int r = 0; r < 8; ++r) {
        const int m = m0 + wm * 32 + mt * 16 + 8 * half + r;  // token index
        const int b = m >> 11, l = m & (L_SEQ - 1);
        Out[(((size_t)(b * NHEADS + h)) * L_SEQ + l) * DHEAD + dh] =
            (bf16_t)(acc[mt][nt][r] + bv);
      }
    }
  }
}

// ---------------------------------------------------------------------------
// Kernel 2: flash attention over one (b,h), q-tile of 64 rows, K/V chunks of 32.
// 4 waves; each wave owns 16 q-rows; Q fragments live in registers.
// K chunk copied with GLOBAL_LOAD_ASYNC_TO_LDS_B128 (ASYNCcnt).
// ---------------------------------------------------------------------------
__global__ __launch_bounds__(128) void flash_attn_kernel(
    const bf16_t* __restrict__ Qh, const bf16_t* __restrict__ Kh,
    const bf16_t* __restrict__ Vh, const unsigned char* __restrict__ amask,
    const float* __restrict__ head_mask, bf16_t* __restrict__ Ctx)
{
  constexpr int LDK = 72;  // K rows: 64 + 8 pad
  constexpr int LDV = 40;  // Vt rows: 32 + 8 pad
  constexpr int LDP = 40;
  __shared__ bf16_t Ks [32 * LDK];
  __shared__ bf16_t Vts[64 * LDV];
  __shared__ bf16_t Ps [4][16 * LDP];

  const int tid  = threadIdx.x;
  const int lane = tid & 31;
  const int wave = tid >> 5;
  const int bh   = blockIdx.x;          // 0..31
  const int q0   = blockIdx.y * 64;
  const int b    = bh >> 4, h = bh & 15;
  const int nl   = lane & 15, half = lane >> 4;

  const bf16_t* Qbh = Qh + (size_t)bh * L_SEQ * DHEAD;
  const bf16_t* Kbh = Kh + (size_t)bh * L_SEQ * DHEAD;
  const bf16_t* Vbh = Vh + (size_t)bh * L_SEQ * DHEAD;
  const unsigned char* mb = amask + (size_t)b * L_SEQ * L_SEQ;

  // Q A-fragments in registers (16 rows x 64 -> two 16x32 fragments)
  v16bf aq[2];
  {
    const bf16_t* qp = Qbh + (size_t)(q0 + wave * 16 + nl) * DHEAD;
    FragU f0, f1;
    f0.q[0] = *reinterpret_cast<const f32x4*>(qp + 8 * half);
    f0.q[1] = *reinterpret_cast<const f32x4*>(qp + 16 + 8 * half);
    f1.q[0] = *reinterpret_cast<const f32x4*>(qp + 32 + 8 * half);
    f1.q[1] = *reinterpret_cast<const f32x4*>(qp + 48 + 8 * half);
    aq[0] = f0.v; aq[1] = f1.v;
  }

  float mrun[8], lrun[8];
  v8f accO[4];
#pragma unroll
  for (int r = 0; r < 8; ++r) { mrun[r] = -3.0e38f; lrun[r] = 0.0f; }
#pragma unroll
  for (int dt = 0; dt < 4; ++dt) accO[dt] = (v8f){};

  const int qrow_base = q0 + wave * 16 + 8 * half;

  for (int kc = 0; kc < L_SEQ; kc += 32) {
    __syncthreads();
    // K chunk [32][64] -> Ks row-major (async); V chunk transposed -> Vts[d][j]
#pragma unroll
    for (int i = 0; i < 2; ++i) {
      const int idx = tid + i * 128;  // 0..255
      const int r   = idx >> 3;       // 0..31  (k-row j)
      const int c   = (idx & 7) * 8;  // 0,8,..,56 (d)
      const bf16_t* gk = Kbh + (size_t)(kc + r) * DHEAD + c;
#if HAVE_ASYNC_LDS
      async_cp_b128(&Ks[r * LDK + c], gk);
#else
      *reinterpret_cast<f32x4*>(&Ks[r * LDK + c]) =
          *reinterpret_cast<const f32x4*>(gk);
#endif
      Pack8 pv;
      pv.q = *reinterpret_cast<const f32x4*>(Vbh + (size_t)(kc + r) * DHEAD + c);
#pragma unroll
      for (int e = 0; e < 8; ++e) Vts[(c + e) * LDV + r] = pv.e[e];
    }
    // prefetch next chunk (one 128B line per row)
    const int nk = kc + 32;
    if (nk < L_SEQ) {
      if (tid < 32)      __builtin_prefetch(Kbh + (size_t)(nk + tid) * DHEAD, 0, 0);
      else if (tid < 64) __builtin_prefetch(Vbh + (size_t)(nk + tid - 32) * DHEAD, 0, 0);
    }
    async_wait0();
    __syncthreads();

    // S = (Q K^T) * 0.125, masked
    v8f s[2];
#pragma unroll
    for (int jt = 0; jt < 2; ++jt) {
      v16bf b0 = load_b_frag<LDK>(&Ks[jt * 16 * LDK], lane);       // d 0..31
      v16bf b1 = load_b_frag<LDK>(&Ks[jt * 16 * LDK + 32], lane);  // d 32..63
      v8f z = (v8f){};
      z = wmma_bf16(aq[0], b0, z);
      s[jt] = wmma_bf16(aq[1], b1, z);
    }
#pragma unroll
    for (int jt = 0; jt < 2; ++jt) {
      const int col = kc + jt * 16 + nl;
#pragma unroll
      for (int r = 0; r < 8; ++r) {
        float sv = s[jt][r] * 0.125f;
        if (mb[(size_t)(qrow_base + r) * L_SEQ + col]) sv = NEGBIG;
        s[jt][r] = sv;
      }
    }

    // online softmax (row reductions across the 16-lane half-group)
#pragma unroll
    for (int r = 0; r < 8; ++r) {
      float mx = fmaxf(s[0][r], s[1][r]);
#pragma unroll
      for (int off = 8; off >= 1; off >>= 1)
        mx = fmaxf(mx, __shfl_xor(mx, off, 32));
      const float mnew  = fmaxf(mrun[r], mx);
      const float alpha = __expf(mrun[r] - mnew);
      const float p0 = __expf(s[0][r] - mnew);
      const float p1 = __expf(s[1][r] - mnew);
      float rs = p0 + p1;
#pragma unroll
      for (int off = 8; off >= 1; off >>= 1)
        rs += __shfl_xor(rs, off, 32);
      lrun[r] = lrun[r] * alpha + rs;
      mrun[r] = mnew;
      s[0][r] = p0; s[1][r] = p1;
#pragma unroll
      for (int dt = 0; dt < 4; ++dt) accO[dt][r] *= alpha;
    }

    // stage P (D-layout) through wave-private LDS, reload as A-fragment
    bf16_t* pw = Ps[wave];
#pragma unroll
    for (int jt = 0; jt < 2; ++jt)
#pragma unroll
      for (int r = 0; r < 8; ++r)
        pw[(8 * half + r) * LDP + jt * 16 + nl] = (bf16_t)s[jt][r];
    v16bf ap = load_a_frag<LDP>(pw, lane);  // same-wave LDS ops are in order

    // O += P x Vchunk
#pragma unroll
    for (int dt = 0; dt < 4; ++dt) {
      v16bf bv = load_b_frag<LDV>(&Vts[dt * 16 * LDV], lane);
      accO[dt] = wmma_bf16(ap, bv, accO[dt]);
    }
  }

  // epilogue: O / l * head_mask -> Ctx bf16 [B,L,D] at column h*64
  const float hm = head_mask[0];
#pragma unroll
  for (int r = 0; r < 8; ++r) {
    const float inv  = hm / lrun[r];
    const int   qrow = qrow_base + r;
    bf16_t* cp = Ctx + ((size_t)(b * L_SEQ + qrow)) * D_MODEL + h * DHEAD;
#pragma unroll
    for (int dt = 0; dt < 4; ++dt)
      cp[dt * 16 + nl] = (bf16_t)(accO[dt][r] * inv);
  }
}

// ---------------------------------------------------------------------------
// Kernel 3: out = Ctx(bf16) @ Wo^T + bo + q   (fp32 out, residual)
// A tile (already bf16) copied with async-to-LDS.
// ---------------------------------------------------------------------------
__global__ __launch_bounds__(256) void out_gemm_kernel(
    const bf16_t* __restrict__ A,    // [4096,1024] bf16
    const float*  __restrict__ W,    // [1024,1024]
    const float*  __restrict__ bias, // [1024]
    const float*  __restrict__ Res,  // [4096,1024] residual (q input)
    float* __restrict__ Out)         // [4096,1024]
{
  constexpr int LDT = 40;
  __shared__ bf16_t As[128 * LDT];
  __shared__ bf16_t Bs[128 * LDT];

  const int tid  = threadIdx.x;
  const int lane = tid & 31;
  const int wave = tid >> 5;
  const int wm   = wave & 3;
  const int wn   = wave >> 2;
  const int m0   = blockIdx.x * 128;
  const int n0   = blockIdx.y * 128;

  v8f acc[2][4];
#pragma unroll
  for (int i = 0; i < 2; ++i)
#pragma unroll
    for (int j = 0; j < 4; ++j) acc[i][j] = (v8f){};

  for (int k0 = 0; k0 < D_MODEL; k0 += 32) {
    // A tile is already bf16: async 16B copies
#pragma unroll
    for (int i = 0; i < 2; ++i) {
      const int idx = tid + i * 256;  // 0..511
      const int r   = idx >> 2;       // 0..127
      const int c   = (idx & 3) * 8;  // 0,8,16,24
      const bf16_t* ga = A + (size_t)(m0 + r) * D_MODEL + k0 + c;
#if HAVE_ASYNC_LDS
      async_cp_b128(&As[r * LDT + c], ga);
#else
      *reinterpret_cast<f32x4*>(&As[r * LDT + c]) =
          *reinterpret_cast<const f32x4*>(ga);
#endif
    }
    // W tile: fp32 -> bf16
#pragma unroll
    for (int i = 0; i < 4; ++i) {
      const int idx = tid + i * 256;
      const int r   = idx >> 3;
      const int c   = (idx & 7) * 4;
      f32x4 b4 = *reinterpret_cast<const f32x4*>(W + (size_t)(n0 + r) * D_MODEL + k0 + c);
      Pack4 pb;
#pragma unroll
      for (int e = 0; e < 4; ++e) pb.e[e] = (bf16_t)b4[e];
      *reinterpret_cast<unsigned long long*>(&Bs[r * LDT + c]) = pb.u;
    }
    const int pk = k0 + 32;
    if (pk < D_MODEL) {
      if (tid < 128) __builtin_prefetch(A + (size_t)(m0 + tid) * D_MODEL + pk, 0, 0);
      else           __builtin_prefetch(W + (size_t)(n0 + tid - 128) * D_MODEL + pk, 0, 0);
    }
    async_wait0();
    __syncthreads();

    v16bf af[2], bfv[4];
#pragma unroll
    for (int mt = 0; mt < 2; ++mt)
      af[mt] = load_a_frag<LDT>(&As[(wm * 32 + mt * 16) * LDT], lane);
#pragma unroll
    for (int nt = 0; nt < 4; ++nt)
      bfv[nt] = load_b_frag<LDT>(&Bs[(wn * 64 + nt * 16) * LDT], lane);
#pragma unroll
    for (int mt = 0; mt < 2; ++mt)
#pragma unroll
      for (int nt = 0; nt < 4; ++nt)
        acc[mt][nt] = wmma_bf16(af[mt], bfv[nt], acc[mt][nt]);
    __syncthreads();
  }

  const int nl = lane & 15, half = lane >> 4;
#pragma unroll
  for (int mt = 0; mt < 2; ++mt) {
#pragma unroll
    for (int nt = 0; nt < 4; ++nt) {
      const int o  = n0 + wn * 64 + nt * 16 + nl;
      const float bv = bias[o];
#pragma unroll
      for (int r = 0; r < 8; ++r) {
        const int m = m0 + wm * 32 + mt * 16 + 8 * half + r;
        const size_t off = (size_t)m * D_MODEL + o;
        Out[off] = acc[mt][nt][r] + bv + Res[off];
      }
    }
  }
}

// ---------------------------------------------------------------------------
extern "C" void kernel_launch(void* const* d_in, const int* in_sizes, int n_in,
                              void* d_out, int out_size, void* d_ws, size_t ws_size,
                              hipStream_t stream) {
  const float* q  = (const float*)d_in[0];
  const float* k  = (const float*)d_in[1];
  const float* v  = (const float*)d_in[2];
  const unsigned char* amask = (const unsigned char*)d_in[3];  // bool [B,L,L]
  const float* head_mask = (const float*)d_in[4];              // broadcast scalar
  const float* Wq = (const float*)d_in[5];
  const float* bq = (const float*)d_in[6];
  const float* Wk = (const float*)d_in[7];
  const float* bk = (const float*)d_in[8];
  const float* Wv = (const float*)d_in[9];
  const float* bv = (const float*)d_in[10];
  const float* Wo = (const float*)d_in[11];
  const float* bo = (const float*)d_in[12];
  float* out = (float*)d_out;

  char* ws = (char*)d_ws;
  bf16_t* Qh  = (bf16_t*)(ws);
  bf16_t* Kh  = (bf16_t*)(ws + ((size_t)8  << 20));
  bf16_t* Vh  = (bf16_t*)(ws + ((size_t)16 << 20));
  bf16_t* Ctx = (bf16_t*)(ws + ((size_t)24 << 20));

  dim3 gGemm(4096 / 128, 1024 / 128);  // 32 x 8
  qkv_gemm_kernel<<<gGemm, 256, 0, stream>>>(q, Wq, bq, Qh);
  qkv_gemm_kernel<<<gGemm, 256, 0, stream>>>(k, Wk, bk, Kh);
  qkv_gemm_kernel<<<gGemm, 256, 0, stream>>>(v, Wv, bv, Vh);

  dim3 gAttn(32, 2048 / 64);           // (B*H) x q-tiles
  flash_attn_kernel<<<gAttn, 128, 0, stream>>>(Qh, Kh, Vh, amask, head_mask, Ctx);

  out_gemm_kernel<<<gGemm, 256, 0, stream>>>(Ctx, Wo, bo, q, out);
}
